// ContextEncoderLayer_29678224015884
// MI455X (gfx1250) — matrix operand order repaired and checked by reading, not compiled, not run on hardware
//
#include <hip/hip_runtime.h>
#include <cmath>
#include <cstdint>

typedef __bf16 bf16_t;
typedef __attribute__((ext_vector_type(16))) __bf16 v16bf;
typedef __attribute__((ext_vector_type(8)))  __bf16 v8bf;
typedef __attribute__((ext_vector_type(8)))  float  v8f;

#define L_  2048
#define D_  32
#define DM_ 1024
#define H_  16
#define FF_ 4096
#define DH_ 64

// ---------------------------------------------------------------------------
// CDNA5 async helpers: GLOBAL_LOAD_ASYNC_TO_LDS_* copy memory -> LDS without
// touching VGPRs, tracked by ASYNCcnt (ISA 10.x / 15.18.3).
// Instruction offset is added to BOTH the global and the LDS address, so one
// address pair serves a 32B copy via offset:0 / offset:16.
// ---------------------------------------------------------------------------
__device__ __forceinline__ void async_copy32B(void* lds_ptr, const void* gptr) {
  const uint32_t lds_off = (uint32_t)(uintptr_t)lds_ptr;  // flat->LDS: addr[31:0]
  asm volatile("global_load_async_to_lds_b128 %0, %1, off\n\t"
               "global_load_async_to_lds_b128 %0, %1, off offset:16"
               :: "v"(lds_off), "v"(gptr) : "memory");
}
__device__ __forceinline__ void async_copy16B(void* lds_ptr, const void* gptr) {
  const uint32_t lds_off = (uint32_t)(uintptr_t)lds_ptr;
  asm volatile("global_load_async_to_lds_b128 %0, %1, off"
               :: "v"(lds_off), "v"(gptr) : "memory");
}
__device__ __forceinline__ void wait_async0() {
  asm volatile("s_wait_asynccnt 0x0" ::: "memory");
}

// ---------------------------------------------------------------------------
// Weight transpose + f32 -> bf16 convert:  out[c*R + r] = (bf16) in[r*C + c]
// Makes B operands N-major so WMMA B fragments are K-contiguous 16B LDS reads.
// ---------------------------------------------------------------------------
__global__ __launch_bounds__(1024) void k_transpose_bf16(
    const float* __restrict__ in, bf16_t* __restrict__ out, int R, int C) {
  __shared__ float tile[32][33];
  const int tx = threadIdx.x, ty = threadIdx.y;
  const int r0 = blockIdx.y * 32, c0 = blockIdx.x * 32;
  tile[ty][tx] = in[(size_t)(r0 + ty) * C + (c0 + tx)];
  __syncthreads();
  out[(size_t)(c0 + ty) * R + (r0 + tx)] = (bf16_t)tile[tx][ty];
}

// ---------------------------------------------------------------------------
// bf16 WMMA GEMM core: C[M,N] = A[M,K] (f32, converted on the fly) @ B[K,N]
// B given transposed/converted as Bt[N,K] bf16. 256 threads, 128x128 tile,
// BK=32, double-buffered LDS. A staged via VGPRs (needs f32->bf16 cvt);
// B staged with global_load_async_to_lds_b128 (ASYNCcnt). One barrier/iter.
// Wave tile 32x64 -> 2x4 v_wmma_f32_16x16x32_bf16 accumulators.
// ---------------------------------------------------------------------------
template<bool OUT_BF16, bool GELU>
__device__ __forceinline__ void gemm_body(
    const float* __restrict__ A, const bf16_t* __restrict__ Bt,
    const float* __restrict__ bias, void* __restrict__ Cv,
    int M, int N, int K, int m0, int n0) {
  constexpr int BK = 32;
  constexpr int LDSS = BK + 8;              // padded bf16 row stride (80B)
  __shared__ bf16_t sA[2][128 * LDSS];
  __shared__ bf16_t sB[2][128 * LDSS];

  const int t    = threadIdx.x;             // 0..255
  const int lane = t & 31;
  const int wave = t >> 5;                  // 8 waves
  const int wm   = wave & 3;                // 4 waves down  (32 rows each)
  const int wn   = wave >> 2;               // 2 waves across (64 cols each)
  const int lrow  = lane & 15;
  const int lhalf = lane >> 4;

  const int row = t >> 1;                   // 0..127 : tile row this thread stages
  const int seg = t & 1;                    // 16-element K segment

  v8f acc[2][4] = {};
  float4 ar[4];

  auto issueB = [&](int buf, int kt) {
    const bf16_t* bp = Bt + (size_t)(n0 + row) * K + kt * BK + seg * 16;
    async_copy32B(&sB[buf][row * LDSS + seg * 16], bp);
  };
  auto loadA = [&](int kt) {
    const float* ap = A + (size_t)(m0 + row) * K + kt * BK + seg * 16;
    ar[0] = ((const float4*)ap)[0];
    ar[1] = ((const float4*)ap)[1];
    ar[2] = ((const float4*)ap)[2];
    ar[3] = ((const float4*)ap)[3];
  };
  auto storeA = [&](int buf) {
    v8bf p0, p1;
    p0[0]=(bf16_t)ar[0].x; p0[1]=(bf16_t)ar[0].y; p0[2]=(bf16_t)ar[0].z; p0[3]=(bf16_t)ar[0].w;
    p0[4]=(bf16_t)ar[1].x; p0[5]=(bf16_t)ar[1].y; p0[6]=(bf16_t)ar[1].z; p0[7]=(bf16_t)ar[1].w;
    p1[0]=(bf16_t)ar[2].x; p1[1]=(bf16_t)ar[2].y; p1[2]=(bf16_t)ar[2].z; p1[3]=(bf16_t)ar[2].w;
    p1[4]=(bf16_t)ar[3].x; p1[5]=(bf16_t)ar[3].y; p1[6]=(bf16_t)ar[3].z; p1[7]=(bf16_t)ar[3].w;
    bf16_t* da = &sA[buf][row * LDSS + seg * 16];
    *(v8bf*)(da)     = p0;
    *(v8bf*)(da + 8) = p1;
  };

  // ---- prologue: fill buffer 0 ----
  issueB(0, 0);
  loadA(0);
  storeA(0);
  wait_async0();
  __syncthreads();

  const int KT = K / BK;
  for (int kt = 0; kt < KT; ++kt) {
    const int buf = kt & 1;
    const bool more = (kt + 1 < KT);
    if (more) { issueB(buf ^ 1, kt + 1); loadA(kt + 1); }

    // ---- WMMA fragment loads (ISA 7.12.2 16-bit layouts) + MACs ----
    v16bf af[2];
    #pragma unroll
    for (int mf = 0; mf < 2; ++mf) {
      const bf16_t* p = &sA[buf][(wm * 32 + mf * 16 + lrow) * LDSS + 8 * lhalf];
      v8bf lo = *(const v8bf*)(p);          // K = 8*half .. +7
      v8bf hi = *(const v8bf*)(p + 16);     // K = 16+8*half .. +7
      af[mf] = __builtin_shufflevector(lo, hi, 0,1,2,3,4,5,6,7,8,9,10,11,12,13,14,15);
    }
    v16bf bfg[4];
    #pragma unroll
    for (int nf = 0; nf < 4; ++nf) {
      const bf16_t* p = &sB[buf][(wn * 64 + nf * 16 + lrow) * LDSS + 8 * lhalf];
      v8bf lo = *(const v8bf*)(p);          // K = 8*half .. +7
      v8bf hi = *(const v8bf*)(p + 16);     // K = 16+8*half .. +7
      bfg[nf] = __builtin_shufflevector(lo, hi, 0,1,2,3,4,5,6,7,8,9,10,11,12,13,14,15);
    }
    #pragma unroll
    for (int mf = 0; mf < 2; ++mf)
      #pragma unroll
      for (int nf = 0; nf < 4; ++nf)
        acc[mf][nf] = __builtin_amdgcn_wmma_f32_16x16x32_bf16(
            false, af[mf], false, bfg[nf], (short)0, acc[mf][nf], false, false);

    if (more) {
      storeA(buf ^ 1);                      // write-side: only buf^1 touched
      wait_async0();                        // B bytes resident before barrier
    }
    __syncthreads();
  }

  // ---- epilogue: bias (+GELU) (+cvt) ----
  #pragma unroll
  for (int mf = 0; mf < 2; ++mf) {
    #pragma unroll
    for (int nf = 0; nf < 4; ++nf) {
      const int n = n0 + wn * 64 + nf * 16 + lrow;   // C/D: N = lane%16
      const float bv = bias[n];
      const int mb = m0 + wm * 32 + mf * 16 + 8 * lhalf; // M = 8*(lane/16)+v
      #pragma unroll
      for (int v = 0; v < 8; ++v) {
        float val = acc[mf][nf][v] + bv;
        if (GELU) val = 0.5f * val * (1.0f + erff(val * 0.70710678118654752f));
        if (OUT_BF16)
          ((bf16_t*)Cv)[(size_t)(mb + v) * N + n] = (bf16_t)val;
        else
          ((float*)Cv)[(size_t)(mb + v) * N + n] = val;
      }
    }
  }
}

template<bool OUT_BF16, bool GELU>
__global__ __launch_bounds__(256) void k_gemm(
    const float* __restrict__ A, const bf16_t* __restrict__ Bt,
    const float* __restrict__ bias, void* __restrict__ C, int M, int N, int K) {
  gemm_body<OUT_BF16, GELU>(A, Bt, bias, C, M, N, K, blockIdx.y * 128, blockIdx.x * 128);
}

// K and V projections fused on grid.z so concurrent blocks share A (target)
// through L2: ~one HBM pass over the 268MB activation instead of two.
__global__ __launch_bounds__(256) void k_gemm_kv(
    const float* __restrict__ A,
    const bf16_t* __restrict__ BtK, const bf16_t* __restrict__ BtV,
    const float* __restrict__ bk, const float* __restrict__ bv,
    bf16_t* __restrict__ CK, bf16_t* __restrict__ CV, int M, int N, int K) {
  const bf16_t* Bt = blockIdx.z ? BtV : BtK;
  const float* bias = blockIdx.z ? bv : bk;
  bf16_t* C = blockIdx.z ? CV : CK;
  gemm_body<true, false>(A, Bt, bias, (void*)C, M, N, K, blockIdx.y * 128, blockIdx.x * 128);
}

// ---------------------------------------------------------------------------
// Per-position attention + residual + LayerNorm1. One workgroup per l.
// K_l / V_l rows (128KB) staged via async LDS copies (~140KB of 320KB pool).
// ---------------------------------------------------------------------------
__global__ __launch_bounds__(256) void k_attn_ln1(
    const float* __restrict__ q, const bf16_t* __restrict__ Km,
    const bf16_t* __restrict__ Vm, const float* __restrict__ src,
    const float* __restrict__ mask, const float* __restrict__ g1,
    const float* __restrict__ beta1, float* __restrict__ xout) {
  const int l = blockIdx.x;
  const int t = threadIdx.x;                 // 256
  __shared__ float  qs[DM_];
  __shared__ bf16_t Ks[D_ * DM_];
  __shared__ bf16_t Vs[D_ * DM_];
  __shared__ float  sc[H_ * D_];
  __shared__ float  xrow[DM_];
  __shared__ float  r1[256];
  __shared__ float  r2[256];

  {
    const bf16_t* kp = Km + (size_t)l * D_ * DM_;
    const bf16_t* vp = Vm + (size_t)l * D_ * DM_;
    #pragma unroll
    for (int i = 0; i < 16; ++i) {           // 16B x 16 x 256thr = 64KB each
      const int e = (t + i * 256) * 8;       // bf16 elements
      async_copy16B(&Ks[e], kp + e);
      async_copy16B(&Vs[e], vp + e);
    }
  }
  ((float4*)qs)[t] = ((const float4*)(q + (size_t)l * DM_))[t];
  wait_async0();
  __syncthreads();

  // scores[h][k] = (q_lh . k_lkh)/sqrt(64) + mask[l][k]
  #pragma unroll
  for (int j = 0; j < 2; ++j) {
    const int p = t + j * 256;               // 0..511
    const int h = p >> 5, k = p & 31;
    const float* qr = &qs[h * DH_];
    const bf16_t* kr = &Ks[k * DM_ + h * DH_];
    float s = 0.0f;
    #pragma unroll
    for (int d = 0; d < DH_; ++d) s += qr[d] * (float)kr[d];
    sc[p] = s * 0.125f + mask[(size_t)l * D_ + k];
  }
  __syncthreads();

  if (t < H_) {                              // softmax over the 32 candidates
    float* rowp = &sc[t * D_];
    float mx = rowp[0];
    #pragma unroll
    for (int k = 1; k < D_; ++k) mx = fmaxf(mx, rowp[k]);
    float sum = 0.0f;
    #pragma unroll
    for (int k = 0; k < D_; ++k) { float e = __expf(rowp[k] - mx); rowp[k] = e; sum += e; }
    const float inv = 1.0f / sum;
    #pragma unroll
    for (int k = 0; k < D_; ++k) rowp[k] *= inv;
  }
  __syncthreads();

  // ctx + residual, accumulate LN statistics
  float ps = 0.0f, pq = 0.0f;
  #pragma unroll
  for (int j = 0; j < 4; ++j) {
    const int idx = t + j * 256;             // 0..1023
    const int h = idx >> 6;
    float c = 0.0f;
    #pragma unroll
    for (int k = 0; k < D_; ++k) c += sc[h * D_ + k] * (float)Vs[k * DM_ + idx];
    const float val = src[(size_t)l * DM_ + idx] + c;
    xrow[idx] = val; ps += val; pq += val * val;
  }
  r1[t] = ps; r2[t] = pq;
  for (int s = 128; s > 0; s >>= 1) {
    __syncthreads();
    if (t < s) { r1[t] += r1[t + s]; r2[t] += r2[t + s]; }
  }
  __syncthreads();
  const float mu = r1[0] * (1.0f / DM_);
  const float var = r2[0] * (1.0f / DM_) - mu * mu;
  const float rstd = rsqrtf(var + 1e-5f);
  #pragma unroll
  for (int j = 0; j < 4; ++j) {
    const int idx = t + j * 256;
    xout[(size_t)l * DM_ + idx] = (xrow[idx] - mu) * rstd * g1[idx] + beta1[idx];
  }
}

// ---------------------------------------------------------------------------
// Final residual + LayerNorm2
// ---------------------------------------------------------------------------
__global__ __launch_bounds__(256) void k_ln2(
    const float* __restrict__ x, const float* __restrict__ ff,
    const float* __restrict__ g2, const float* __restrict__ beta2,
    float* __restrict__ out) {
  const int l = blockIdx.x;
  const int t = threadIdx.x;
  __shared__ float row[DM_];
  __shared__ float r1[256];
  __shared__ float r2[256];
  float ps = 0.0f, pq = 0.0f;
  #pragma unroll
  for (int j = 0; j < 4; ++j) {
    const int idx = t + j * 256;
    const float v = x[(size_t)l * DM_ + idx] + ff[(size_t)l * DM_ + idx];
    row[idx] = v; ps += v; pq += v * v;
  }
  r1[t] = ps; r2[t] = pq;
  for (int s = 128; s > 0; s >>= 1) {
    __syncthreads();
    if (t < s) { r1[t] += r1[t + s]; r2[t] += r2[t + s]; }
  }
  __syncthreads();
  const float mu = r1[0] * (1.0f / DM_);
  const float var = r2[0] * (1.0f / DM_) - mu * mu;
  const float rstd = rsqrtf(var + 1e-5f);
  #pragma unroll
  for (int j = 0; j < 4; ++j) {
    const int idx = t + j * 256;
    out[(size_t)l * DM_ + idx] = (row[idx] - mu) * rstd * g2[idx] + beta2[idx];
  }
}

// ---------------------------------------------------------------------------
extern "C" void kernel_launch(void* const* d_in, const int* in_sizes, int n_in,
                              void* d_out, int out_size, void* d_ws, size_t ws_size,
                              hipStream_t stream) {
  const float* src    = (const float*)d_in[0];
  const float* target = (const float*)d_in[1];
  const float* amask  = (const float*)d_in[2];
  const float* Wq = (const float*)d_in[3];  const float* bq = (const float*)d_in[4];
  const float* Wk = (const float*)d_in[5];  const float* bk = (const float*)d_in[6];
  const float* Wv = (const float*)d_in[7];  const float* bv = (const float*)d_in[8];
  const float* W1 = (const float*)d_in[9];  const float* b1 = (const float*)d_in[10];
  const float* W2 = (const float*)d_in[11]; const float* b2 = (const float*)d_in[12];
  const float* g1 = (const float*)d_in[13]; const float* be1 = (const float*)d_in[14];
  const float* g2 = (const float*)d_in[15]; const float* be2 = (const float*)d_in[16];
  float* out = (float*)d_out;

  char* base = (char*)d_ws;
  size_t off = 0;
  auto carve = [&](size_t bytes) -> char* {
    char* p = base + off; off += (bytes + 255) & ~(size_t)255; return p;
  };
  bf16_t* wqT = (bf16_t*)carve((size_t)DM_ * DM_ * 2);
  bf16_t* wkT = (bf16_t*)carve((size_t)DM_ * DM_ * 2);
  bf16_t* wvT = (bf16_t*)carve((size_t)DM_ * DM_ * 2);
  bf16_t* w1T = (bf16_t*)carve((size_t)DM_ * FF_ * 2);
  bf16_t* w2T = (bf16_t*)carve((size_t)FF_ * DM_ * 2);
  float*  qf  = (float*) carve((size_t)L_ * DM_ * 4);
  bf16_t* km  = (bf16_t*)carve((size_t)L_ * D_ * DM_ * 2);
  bf16_t* vm  = (bf16_t*)carve((size_t)L_ * D_ * DM_ * 2);
  float*  xf  = (float*) carve((size_t)L_ * DM_ * 4);
  float*  h1  = (float*) carve((size_t)L_ * FF_ * 4);
  float*  ffb = (float*) carve((size_t)L_ * DM_ * 4);

  const dim3 tb(32, 32);
  k_transpose_bf16<<<dim3(DM_ / 32, DM_ / 32), tb, 0, stream>>>(Wq, wqT, DM_, DM_);
  k_transpose_bf16<<<dim3(DM_ / 32, DM_ / 32), tb, 0, stream>>>(Wk, wkT, DM_, DM_);
  k_transpose_bf16<<<dim3(DM_ / 32, DM_ / 32), tb, 0, stream>>>(Wv, wvT, DM_, DM_);
  k_transpose_bf16<<<dim3(FF_ / 32, DM_ / 32), tb, 0, stream>>>(W1, w1T, DM_, FF_);
  k_transpose_bf16<<<dim3(DM_ / 32, FF_ / 32), tb, 0, stream>>>(W2, w2T, FF_, DM_);

  // Q = src @ Wq + bq                     (2048 x 1024 x 1024)
  k_gemm<false, false><<<dim3(DM_ / 128, L_ / 128), 256, 0, stream>>>(
      src, wqT, bq, (void*)qf, L_, DM_, DM_);
  // K,V = target @ {Wk,Wv} + {bk,bv}     (65536 x 1024 x 1024, fused on z)
  k_gemm_kv<<<dim3(DM_ / 128, (L_ * D_) / 128, 2), 256, 0, stream>>>(
      target, wkT, wvT, bk, bv, km, vm, L_ * D_, DM_, DM_);
  // attention + residual + LN1
  k_attn_ln1<<<L_, 256, 0, stream>>>(qf, km, vm, src, amask, g1, be1, xf);
  // FFN1 with exact-erf GELU epilogue    (2048 x 4096 x 1024)
  k_gemm<false, true><<<dim3(FF_ / 128, L_ / 128), 256, 0, stream>>>(
      xf, w1T, b1, (void*)h1, L_, FF_, DM_);
  // FFN2                                  (2048 x 1024 x 4096)
  k_gemm<false, false><<<dim3(DM_ / 128, L_ / 128), 256, 0, stream>>>(
      h1, w2T, b2, (void*)ffb, L_, DM_, FF_);
  // residual + LN2
  k_ln2<<<L_, 256, 0, stream>>>(xf, ffb, g2, be2, out);

  (void)in_sizes; (void)n_in; (void)out_size; (void)ws_size;
}
